// GeneralizedHyperConnections_47193100648857
// MI455X (gfx1250) — matrix operand order repaired and checked by reading, not compiled, not run on hardware
//
#include <hip/hip_runtime.h>
#include <hip/hip_bf16.h>

// ---------------------------------------------------------------------------
// GeneralizedHyperConnections on MI455X (gfx1250)
//   rows B = 16384, streams N=4, branches M=2, BLOCK=1024, EMBED=2048
//   Dominant cost: (16384 x 2048) @ W_module^T (2048 x 2048)  ~= 137 GFLOP
//   -> bf16 WMMA (v_wmma_f32_16x16x32_bf16); everything else fp32 VALU.
//   Staging: double-buffered LDS tiles via GLOBAL_LOAD_ASYNC_TO_LDS_B128
//   (next tile's DMA overlaps current tile's WMMAs; s_wait_asynccnt 6).
//   DS->WMMA pipelined via sched_group_barrier so waits are dscnt<=2.
// Workspace layout:
//   [0,        64MB)  : X_read  bf16  (16384 x 2048)
//   [64MB,     72MB)  : W_module bf16 (2048 x 2048)
//   [72MB,   72.5MB)  : wo dynamic   (16384 x 8 f32)
// ---------------------------------------------------------------------------

typedef __attribute__((ext_vector_type(8)))  float          v8f;
typedef __attribute__((ext_vector_type(16))) __bf16         v16bf;
typedef __attribute__((ext_vector_type(16))) unsigned short v16u;

#define ROWS    16384
#define EMBED   2048
#define NBLK    1024
#define ASTR    40      // LDS row stride (bf16 elems), 80B rows keep 16B alignment
#define BSTR    40
#define ABUF_B  (128 * ASTR * 2)        // 10240 B per A buffer
#define BBUF_B  (2 * 128 * BSTR * 2)    // 20480 B per B buffer

static __device__ __forceinline__ unsigned short f2bf(float f) {
  unsigned int u = __builtin_bit_cast(unsigned int, f);
  u += 0x7FFFu + ((u >> 16) & 1u);          // round-to-nearest-even
  return (unsigned short)(u >> 16);
}

// Low 32 bits of a flat pointer to a __shared__ object == LDS byte offset
// (AMDGPU addrspacecast(3->0) puts the aperture in the high half).
static __device__ __forceinline__ unsigned lds_off(const void* p) {
  return (unsigned)(uintptr_t)p;
}

// ---------------------------------------------------------------------------
// Kernel 0: W_module f32 -> bf16 (row-major layout preserved: [out=2048][in=2048])
// ---------------------------------------------------------------------------
__global__ __launch_bounds__(256) void ghc_cvt(const float* __restrict__ Wm,
                                               unsigned short* __restrict__ Wb) {
  size_t idx = ((size_t)blockIdx.x * 256 + threadIdx.x) * 8;
  float4 a = *(const float4*)(Wm + idx);
  float4 b = *(const float4*)(Wm + idx + 4);
  uint4 r;
  r.x = (unsigned)f2bf(a.x) | ((unsigned)f2bf(a.y) << 16);
  r.y = (unsigned)f2bf(a.z) | ((unsigned)f2bf(a.w) << 16);
  r.z = (unsigned)f2bf(b.x) | ((unsigned)f2bf(b.y) << 16);
  r.w = (unsigned)f2bf(b.z) | ((unsigned)f2bf(b.w) << 16);
  *(uint4*)(Wb + idx) = r;
}

// ---------------------------------------------------------------------------
// Kernel 1: per-row prep. One 256-thread block per row (8 wave32).
//   - RMSNorm stats, 8 projections + tanh -> wo/ri/sm
//   - X_read bf16 (GEMM A operand) to ws
//   - residual sm @ xr written straight into d_out
// ---------------------------------------------------------------------------
__global__ __launch_bounds__(256) void ghc_prep(
    const float* __restrict__ x,
    const float* __restrict__ wo_s, const float* __restrict__ ds_wo, const float* __restrict__ W_wo,
    const float* __restrict__ ri_s, const float* __restrict__ ds_ri, const float* __restrict__ W_ri,
    const float* __restrict__ sm_s, const float* __restrict__ ds_sm, const float* __restrict__ W_sm,
    unsigned short* __restrict__ Xb, float* __restrict__ wo_dyn, float* __restrict__ out) {
  __shared__ float xs[4 * NBLK];     // raw row, 16 KB
  __shared__ float psum[2048];       // reductions
  __shared__ float rms_sh[4];
  __shared__ float mix[32];          // [0,8)=wo  [8,16)=ri  [16,32)=sm

  const int t = threadIdx.x;
  const size_t row = blockIdx.x;
  const float* xrow = x + row * 4096;
  const int n = t >> 6;              // stream 0..3
  const int i = t & 63;              // 64 threads per stream

  float loc[16];
  float ss = 0.f;
  const float4* src = (const float4*)(xrow + n * NBLK + i * 16);
#pragma unroll
  for (int q = 0; q < 4; ++q) {
    float4 f = src[q];
    loc[4*q+0] = f.x; loc[4*q+1] = f.y; loc[4*q+2] = f.z; loc[4*q+3] = f.w;
    ss += f.x*f.x + f.y*f.y + f.z*f.z + f.w*f.w;
    *(float4*)(&xs[n * NBLK + i * 16 + 4*q]) = f;
  }
  psum[t] = ss;
  __syncthreads();
#pragma unroll
  for (int off = 32; off >= 1; off >>= 1) {
    if (i < off) psum[t] = psum[t] + psum[t + off];
    __syncthreads();
  }
  if (i == 0)
    rms_sh[n] = rsqrtf(psum[t] * (1.f / 1024.f) + 1.1920928955078125e-07f);
  __syncthreads();

  // 8 projections: p 0..1 -> W_write_out rows, 2..3 -> W_read_in, 4..7 -> W_stream
  float dp[8];
#pragma unroll
  for (int p = 0; p < 8; ++p) dp[p] = 0.f;
  const float* Wp[8] = { W_wo, W_wo + NBLK, W_ri, W_ri + NBLK,
                         W_sm, W_sm + NBLK, W_sm + 2*NBLK, W_sm + 3*NBLK };
  const int dbase = i * 16;
#pragma unroll
  for (int q = 0; q < 16; ++q) {
    float xv = loc[q];
    int d = dbase + q;
#pragma unroll
    for (int p = 0; p < 8; ++p) dp[p] += xv * Wp[p][d];
  }
#pragma unroll
  for (int p = 0; p < 8; ++p) psum[(n * 8 + p) * 64 + i] = dp[p];
  __syncthreads();

  if (t < 32) {                       // one wave finalizes all 32 coefficients
    int n2 = t >> 3, p = t & 7;
    const float* pr = &psum[(n2 * 8 + p) * 64];
    float s = 0.f;
    for (int j2 = 0; j2 < 64; ++j2) s += pr[j2];
    float proj = rms_sh[n2] * s;
    float th = tanhf(proj * 0.03125f);       // / sqrt(1024)
    if (p < 2) {
      float v = ds_wo[n2*2 + p] * th + wo_s[n2*2 + p];
      mix[n2*2 + p] = v;
      wo_dyn[row * 8 + n2*2 + p] = v;
    } else if (p < 4) {
      int m = p - 2;
      mix[8 + n2*2 + m] = ds_ri[n2*2 + m] * th + ri_s[n2*2 + m];
    } else {
      int k2 = p - 4;
      mix[16 + n2*4 + k2] = ds_sm[n2*4 + k2] * th + sm_s[n2*4 + k2];
    }
  }
  __syncthreads();

  // outputs: x_read (bf16) and residual sm @ xr (into d_out)
#pragma unroll
  for (int q = 0; q < 4; ++q) {
    int d = t * 4 + q;
    float x0 = xs[d], x1 = xs[NBLK + d], x2 = xs[2*NBLK + d], x3 = xs[3*NBLK + d];
#pragma unroll
    for (int m = 0; m < 2; ++m) {     // x_read[m][d] = sum_n ri[n][m] * xr[n][d]
      float xr = mix[8 + m] * x0 + mix[8 + 2 + m] * x1 + mix[8 + 4 + m] * x2 + mix[8 + 6 + m] * x3;
      Xb[row * EMBED + m * NBLK + d] = f2bf(xr);
    }
#pragma unroll
    for (int nn = 0; nn < 4; ++nn) {  // residual
      float r = mix[16 + nn*4 + 0] * x0 + mix[16 + nn*4 + 1] * x1 +
                mix[16 + nn*4 + 2] * x2 + mix[16 + nn*4 + 3] * x3;
      out[row * 4096 + nn * NBLK + d] = r;
    }
  }
}

// ---------------------------------------------------------------------------
// Kernel 2: bf16 WMMA GEMM  C = X_read @ W_module^T  with fused wo-epilogue.
// Block: 256 thr = 8 waves. Tile: 128 rows x (128 cols in BOTH halves j, j+1024).
// Wave w owns rows [w*16, w*16+16); per K-step: 1 A frag + 16 B frags,
// 16 x v_wmma_f32_16x16x32_bf16; accumulators 2x8 v8f = 128 VGPRs.
// ---------------------------------------------------------------------------
__global__ __launch_bounds__(256) void ghc_gemm(
    const unsigned short* __restrict__ Xb, const unsigned short* __restrict__ Wb,
    const float* __restrict__ wo_dyn, float* __restrict__ out) {
  __shared__ __align__(16) unsigned short As[2][128 * ASTR];        // 2 x 10 KB
  __shared__ __align__(16) unsigned short Bs[2][2 * 128 * BSTR];    // 2 x 20 KB
  __shared__ float wos[128 * 8];                                    // 4 KB

  const int t = threadIdx.x;
  const int lane = t & 31;
  const int w = t >> 5;
  const int lr = lane & 15;
  const int rt = blockIdx.x;           // row tile (128 rows)
  const int c0 = blockIdx.y * 128;     // column tile inside [0,1024)

  v8f acc[2][8];
#pragma unroll
  for (int h = 0; h < 2; ++h)
#pragma unroll
    for (int ct = 0; ct < 8; ++ct)
#pragma unroll
      for (int e = 0; e < 8; ++e) acc[h][ct][e] = 0.f;

  const unsigned short* Ag = Xb + (size_t)rt * 128 * EMBED;

  // ---- async-staging addresses (LDS offsets fixed; global advances 64B/step)
  unsigned aLds[2], aOff[2];
#pragma unroll
  for (int l = 0; l < 2; ++l) {
    int q = t + l * 256;                    // 0..511
    int r = q >> 2;                         // 0..127
    int kk = (q & 3) << 3;                  // 0,8,16,24 (bf16 elems)
    aLds[l] = lds_off(&As[0][r * ASTR + kk]);
    aOff[l] = (unsigned)((r * EMBED + kk) * 2);
  }
  unsigned bLds[4], bOff[4];
#pragma unroll
  for (int l = 0; l < 4; ++l) {
    int q = t + l * 256;                    // 0..1023
    int h = q >> 9;
    int qq = q & 511;
    int c = qq >> 2;                        // 0..127
    int kk = (qq & 3) << 3;
    bLds[l] = lds_off(&Bs[0][(h * 128 + c) * BSTR + kk]);
    bOff[l] = (unsigned)(((h * NBLK + c0 + c) * EMBED + kk) * 2);
  }

  // issue one K-step tile (6 async b128 per thread) into LDS buffer `sel`
  auto stage = [&](int sel) {
    unsigned aSel = (unsigned)(sel * ABUF_B);
    unsigned bSel = (unsigned)(sel * BBUF_B);
#pragma unroll
    for (int l = 0; l < 2; ++l) {
      asm volatile("global_load_async_to_lds_b128 %0, %1, %2"
                   :: "v"(aLds[l] + aSel), "v"(aOff[l]), "s"(Ag) : "memory");
      aOff[l] += 64;
    }
#pragma unroll
    for (int l = 0; l < 4; ++l) {
      asm volatile("global_load_async_to_lds_b128 %0, %1, %2"
                   :: "v"(bLds[l] + bSel), "v"(bOff[l]), "s"(Wb) : "memory");
      bOff[l] += 64;
    }
  };

  const int ko = (lane < 16) ? 0 : 16;      // A frag K split (contiguous, matches B)
  const int kb = (lane < 16) ? 0 : 16;      // B frag K split

  stage(0);                                 // prologue: tile k=0 -> buffer 0

  for (int k0 = 0; k0 < EMBED; k0 += 32) {
    const int cb = (k0 >> 5) & 1;
    // Overlap: next tile's global->LDS DMA runs under this tile's WMMAs.
    if (k0 + 32 < EMBED) {
      stage(cb ^ 1);                        // buffer read last iter; freed by trailing barrier
      asm volatile("s_wait_asynccnt 0x6" ::: "memory");   // in-order: current tile resident
    } else {
      asm volatile("s_wait_asynccnt 0x0" ::: "memory");
    }
    __syncthreads();                        // all waves' staging visible

    // A fragment (16x32 bf16): lanes 0-15 -> K {0..15}, lanes 16-31 -> {16..31}
    v16u au;
    const unsigned short* ap = &As[cb][(w * 16 + lr) * ASTR];
#pragma unroll
    for (int i2 = 0; i2 < 8; ++i2) { au[i2] = ap[ko + i2]; au[8 + i2] = ap[ko + 8 + i2]; }
    v16bf af = __builtin_bit_cast(v16bf, au);

#pragma unroll
    for (int f = 0; f < 16; ++f) {          // f = h*8 + ct
      int hh = f >> 3, cc = f & 7;
      v16u bu;
      const unsigned short* bp = &Bs[cb][(hh * 128 + cc * 16 + lr) * BSTR + kb];
#pragma unroll
      for (int i2 = 0; i2 < 16; ++i2) bu[i2] = bp[i2];
      v16bf bfr = __builtin_bit_cast(v16bf, bu);
      acc[hh][cc] = __builtin_amdgcn_wmma_f32_16x16x32_bf16(
          false, af, false, bfr, (short)0, acc[hh][cc], false, false);
    }

    // Pin the schedule: DS(Afrag+B0)=4, then 15 x {DS 2 (B_{f+1}), WMMA (B_f)},
    // final WMMA. Forces next-fragment loads ahead of each WMMA -> dscnt<=2,
    // and RA must keep two fragment buffers alive.
    __builtin_amdgcn_sched_group_barrier(0x100, 4, 0);    // DS read x4
#pragma unroll
    for (int f = 0; f < 15; ++f) {
      __builtin_amdgcn_sched_group_barrier(0x100, 2, 0);  // DS read x2
      __builtin_amdgcn_sched_group_barrier(0x008, 1, 0);  // WMMA x1
    }
    __builtin_amdgcn_sched_group_barrier(0x008, 1, 0);    // WMMA x1

    __syncthreads();                        // readers done before buffer is re-staged
  }

#pragma unroll
  for (int l = 0; l < 4; ++l) {             // stage this tile's wo rows
    int idx = t + l * 256;
    wos[idx] = wo_dyn[(size_t)rt * 128 * 8 + idx];
  }
  __syncthreads();

  // Epilogue: out[gr, n, j] += wo[n,0]*C(j) + wo[n,1]*C(j+1024)  (residual already in out)
  // C/D layout: element v -> M = v (lanes 0-15) or 8+v (lanes 16-31), N = lane&15
#pragma unroll
  for (int ct = 0; ct < 8; ++ct) {
    int j = c0 + ct * 16 + lr;
#pragma unroll
    for (int v = 0; v < 8; ++v) {
      int mloc = w * 16 + ((lane < 16) ? v : 8 + v);
      size_t gr = (size_t)rt * 128 + mloc;
      float a0 = acc[0][ct][v];
      float a1 = acc[1][ct][v];
      const float* wrow = &wos[mloc * 8];
#pragma unroll
      for (int nn = 0; nn < 4; ++nn) {
        float* po = out + gr * 4096 + nn * NBLK + j;
        *po += wrow[nn * 2] * a0 + wrow[nn * 2 + 1] * a1;
      }
    }
  }
}

// ---------------------------------------------------------------------------
extern "C" void kernel_launch(void* const* d_in, const int* in_sizes, int n_in,
                              void* d_out, int out_size, void* d_ws, size_t ws_size,
                              hipStream_t stream) {
  const float* x     = (const float*)d_in[0];
  const float* wo_s  = (const float*)d_in[1];
  const float* ds_wo = (const float*)d_in[2];
  const float* W_wo  = (const float*)d_in[3];
  const float* ri_s  = (const float*)d_in[4];
  const float* ds_ri = (const float*)d_in[5];
  const float* W_ri  = (const float*)d_in[6];
  const float* sm_s  = (const float*)d_in[7];
  const float* ds_sm = (const float*)d_in[8];
  const float* W_sm  = (const float*)d_in[9];
  const float* Wm    = (const float*)d_in[10];
  float* out = (float*)d_out;

  const size_t X_BYTES = (size_t)ROWS * EMBED * 2;   // 64 MB
  const size_t W_BYTES = (size_t)EMBED * EMBED * 2;  //  8 MB
  unsigned short* Xb = (unsigned short*)d_ws;
  unsigned short* Wb = (unsigned short*)((char*)d_ws + X_BYTES);
  float* wo_dyn      = (float*)((char*)d_ws + X_BYTES + W_BYTES);

  // 0) W_module -> bf16 (L2-resident, reused by all 16384 rows)
  ghc_cvt<<<(EMBED * EMBED) / (256 * 8), 256, 0, stream>>>(Wm, Wb);
  // 1) per-row norm/tanh/mixing; writes X_read (bf16), wo, and residual into out
  ghc_prep<<<ROWS, 256, 0, stream>>>(x, wo_s, ds_wo, W_wo, ri_s, ds_ri, W_ri,
                                     sm_s, ds_sm, W_sm, Xb, wo_dyn, out);
  // 2) bf16 WMMA GEMM + fused wo epilogue accumulating into out
  ghc_gemm<<<dim3(ROWS / 128, NBLK / 128), 256, 0, stream>>>(Xb, Wb, wo_dyn, out);
}
